// PostProcessCocoGrounding_10960756540240
// MI455X (gfx1250) — matrix operand order repaired and checked by reading, not compiled, not run on hardware
//
#include <hip/hip_runtime.h>
#include <hip/hip_bf16.h>
#include <float.h>

// ---------------- problem constants ----------------
#define BATCH   64
#define QRY     900
#define TOK     256           // K dimension
#define LBL     769

// macro-tile: 32 (M) x 64 (N) per wave => 2x4 WMMA 16x16 tiles
// block = 8 waves = 8 consecutive Q-tiles sharing ONE L-tile (B staged in LDS)
#define QTILE   32
#define LTILE   64
#define QT      29            // ceil(900/32)
#define QPAD    (QT * QTILE)  // 928
#define LT      13            // ceil(769/64)
#define LPAD    (LT * LTILE)  // 832
#define QTG     4             // ceil(QT/8) q-tile groups per L-tile
#define WAVES_PER_BLOCK 8
#define HBINS   1024
#define CAP     4096          // candidate cap per batch
#define B_HALVES (LTILE * TOK)   // 16384 halves = 32KB LDS B tile

typedef __attribute__((ext_vector_type(16))) _Float16 v16h;
typedef __attribute__((ext_vector_type(8)))  _Float16 v8h;
typedef __attribute__((ext_vector_type(8)))  float    v8f;

__constant__ int c_token_idxs[80] = {
    0, 9, 19, 25, 38, 49, 55, 63, 71, 78, 94, 109, 121, 137, 145, 152,
    158, 164, 172, 180, 186, 197, 204, 212, 222, 233, 244, 254, 260, 271,
    281, 288, 300, 314, 321, 336, 353, 366, 378, 394, 403, 416, 422, 429,
    437, 445, 452, 461, 469, 480, 489, 500, 509, 519, 527, 535, 542, 550,
    558, 573, 579, 594, 603, 608, 617, 625, 634, 645, 658, 670, 677, 687,
    694, 709, 716, 724, 731, 742, 755, 768};

// ---------------- prep kernels ----------------

__global__ void k_prep_a(const float* __restrict__ logits, _Float16* __restrict__ A) {
    int idx = blockIdx.x * blockDim.x + threadIdx.x;
    int total = BATCH * QPAD * TOK;
    if (idx >= total) return;
    int t = idx % TOK;
    int q = (idx / TOK) % QPAD;
    int b = idx / (TOK * QPAD);
    float v = 0.0f;
    if (q < QRY) {
        float x = logits[((size_t)b * QRY + q) * TOK + t];
        v = 1.0f / (1.0f + __expf(-x));
    }
    A[idx] = (_Float16)v;
}

__global__ void k_prep_b(const float* __restrict__ pos_map, _Float16* __restrict__ Bm) {
    int idx = blockIdx.x * blockDim.x + threadIdx.x;
    int total = LPAD * TOK;
    if (idx >= total) return;
    int t = idx % TOK;
    int l = idx / TOK;
    float v = (l < LBL) ? pos_map[(size_t)l * TOK + t] : 0.0f;
    Bm[idx] = (_Float16)v;
}

__global__ void k_zero(int* __restrict__ hist, int* __restrict__ cnt) {
    int idx = blockIdx.x * blockDim.x + threadIdx.x;
    if (idx < BATCH * HBINS) hist[idx] = 0;
    if (idx < BATCH) cnt[idx] = 0;
}

// ---------------- shared helpers ----------------

// cooperative load of one 64x256 f16 B macro-tile into LDS (32KB)
__device__ __forceinline__ void load_b_tile(const _Float16* __restrict__ Bm, int lt,
                                            _Float16* sB, int tid) {
    const _Float16* src = Bm + (size_t)lt * LTILE * TOK;
    // 2048 chunks of 8 halves (16B); 256 threads -> 8 chunks each
#pragma unroll
    for (int c = 0; c < 8; ++c) {
        int chunk = tid + c * 256;
        *(v8h*)(sB + chunk * 8) = *(const v8h*)(src + chunk * 8);
    }
}

// fused WMMA macro-tile: 32(M) x 64(N), K=256; B fragments come from LDS
struct Acc { v8f c[2][4]; };

__device__ __forceinline__ Acc gemm_tile(const _Float16* __restrict__ A,
                                         const _Float16* sB,
                                         int b, int qt, int lane) {
    const int mn   = lane & 15;      // M row (A/C) and N col (B/C)
    const int half = lane >> 4;      // lane group
    const int kA   = half * 8;       // A: chunk base (0 or 8), plus +16 chunk
    const int kB   = half * 8;       // B: chunk base (0 or 8), plus +16 chunk

    const _Float16* arow0 = A + ((size_t)(b * QPAD + qt * QTILE + mn)) * TOK;
    const _Float16* arow1 = arow0 + (size_t)16 * TOK;
    const _Float16* brow[4];
#pragma unroll
    for (int j = 0; j < 4; ++j)
        brow[j] = sB + (j * 16 + mn) * TOK + kB;

    Acc acc;
#pragma unroll
    for (int i = 0; i < 2; ++i)
#pragma unroll
        for (int j = 0; j < 4; ++j)
            acc.c[i][j] = (v8f){};

#pragma unroll
    for (int ks = 0; ks < TOK; ks += 32) {
        union { v8h h[2]; v16h v; } a0, a1, bb[4];
        a0.h[0] = *(const v8h*)(arow0 + ks + kA);
        a0.h[1] = *(const v8h*)(arow0 + ks + kA + 16);
        a1.h[0] = *(const v8h*)(arow1 + ks + kA);
        a1.h[1] = *(const v8h*)(arow1 + ks + kA + 16);
#pragma unroll
        for (int j = 0; j < 4; ++j) {
            bb[j].h[0] = *(const v8h*)(brow[j] + ks);      // ds_load_b128
            bb[j].h[1] = *(const v8h*)(brow[j] + ks + 16); // ds_load_b128
        }
#pragma unroll
        for (int j = 0; j < 4; ++j) {
            acc.c[0][j] = __builtin_amdgcn_wmma_f32_16x16x32_f16(
                false, a0.v, false, bb[j].v, (short)0, acc.c[0][j], false, false);
            acc.c[1][j] = __builtin_amdgcn_wmma_f32_16x16x32_f16(
                false, a1.v, false, bb[j].v, (short)0, acc.c[1][j], false, false);
        }
    }
    return acc;
}

// ---------------- pass 1: histogram of prob values ----------------
// grid: (LT * QTG, BATCH); block = 256 = 8 waves (8 consecutive q-tiles)
__global__ void k_hist(const _Float16* __restrict__ A, const _Float16* __restrict__ Bm,
                       int* __restrict__ ghist) {
    __shared__ __align__(16) _Float16 sB[B_HALVES];
    __shared__ int lhist[HBINS];

    const int tid  = threadIdx.x;
    const int b    = blockIdx.y;
    const int lt   = blockIdx.x / QTG;
    const int qtg  = blockIdx.x % QTG;
    const int wave = tid >> 5;
    const int lane = tid & 31;
    const int qt   = qtg * WAVES_PER_BLOCK + wave;

    for (int i = tid; i < HBINS; i += blockDim.x) lhist[i] = 0;
    load_b_tile(Bm, lt, sB, tid);
    __syncthreads();

    if (qt < QT) {
        Acc acc = gemm_tile(A, sB, b, qt, lane);
        const int n = lane & 15, half = lane >> 4;
#pragma unroll
        for (int i = 0; i < 2; ++i) {
#pragma unroll
            for (int j = 0; j < 4; ++j) {
#pragma unroll
                for (int r = 0; r < 8; ++r) {
                    int q   = qt * QTILE + i * 16 + r + half * 8;
                    int lab = lt * LTILE + j * 16 + n;
                    if (q < QRY && lab < LBL) {
                        int bin = (int)(acc.c[i][j][r] * 4.0f);   // prob in [0, 256)
                        bin = bin < 0 ? 0 : (bin > HBINS - 1 ? HBINS - 1 : bin);
                        atomicAdd(&lhist[bin], 1);
                    }
                }
            }
        }
    }
    __syncthreads();
    for (int i = tid; i < HBINS; i += blockDim.x) {
        int v = lhist[i];
        if (v) atomicAdd(&ghist[b * HBINS + i], v);
    }
}

// ---------------- pass 2: per-batch threshold from histogram ----------------
__global__ void k_thresh(const int* __restrict__ ghist, const int* __restrict__ nsel_p,
                         float* __restrict__ thresh) {
    if (threadIdx.x != 0) return;
    int b = blockIdx.x;
    int k = nsel_p[0];
    int cum = 0, bin = 0;
    for (int i = HBINS - 1; i >= 0; --i) {
        cum += ghist[b * HBINS + i];
        if (cum >= k) { bin = i; break; }
    }
    thresh[b] = (float)bin * 0.25f;   // lower edge: guarantees >= k candidates
}

// ---------------- pass 3: recompute + collect candidates ----------------
__global__ void k_collect(const _Float16* __restrict__ A, const _Float16* __restrict__ Bm,
                          const float* __restrict__ thresh, int* __restrict__ cnt,
                          float* __restrict__ candV, int* __restrict__ candI) {
    __shared__ __align__(16) _Float16 sB[B_HALVES];

    const int tid  = threadIdx.x;
    const int b    = blockIdx.y;
    const int lt   = blockIdx.x / QTG;
    const int qtg  = blockIdx.x % QTG;
    const int wave = tid >> 5;
    const int lane = tid & 31;
    const int qt   = qtg * WAVES_PER_BLOCK + wave;

    load_b_tile(Bm, lt, sB, tid);
    __syncthreads();
    if (qt >= QT) return;

    const float th = thresh[b];
    Acc acc = gemm_tile(A, sB, b, qt, lane);
    const int n = lane & 15, half = lane >> 4;
#pragma unroll
    for (int i = 0; i < 2; ++i) {
#pragma unroll
        for (int j = 0; j < 4; ++j) {
#pragma unroll
            for (int r = 0; r < 8; ++r) {
                int q   = qt * QTILE + i * 16 + r + half * 8;
                int lab = lt * LTILE + j * 16 + n;
                if (q < QRY && lab < LBL) {
                    float v = acc.c[i][j][r];
                    if (v >= th) {
                        int p = atomicAdd(&cnt[b], 1);
                        if (p < CAP) {
                            candV[b * CAP + p] = v;
                            candI[b * CAP + p] = q * LBL + lab;   // flat idx
                        }
                    }
                }
            }
        }
    }
}

// ---------------- pass 4: exact top-k selection + outputs ----------------
__global__ void k_select(const float* __restrict__ candV, const int* __restrict__ candI,
                         const int* __restrict__ cnt, const int* __restrict__ nsel_p,
                         const float* __restrict__ bbox, const float* __restrict__ tsize,
                         float* __restrict__ out) {
    __shared__ float sv[CAP];
    __shared__ int   si[CAP];
    __shared__ float rv[256];
    __shared__ int   ri[256];   // global flat idx (tie-break)
    __shared__ int   rs[256];   // slot (for invalidation)

    const int b   = blockIdx.x;
    const int tid = threadIdx.x;
    const int k   = nsel_p[0];
    int n = cnt[b]; if (n > CAP) n = CAP;

    for (int i = tid; i < n; i += blockDim.x) {
        sv[i] = candV[b * CAP + i];
        si[i] = candI[b * CAP + i];
    }
    __syncthreads();

    for (int sel = 0; sel < k; ++sel) {
        float bv = -FLT_MAX; int bi = 0x7fffffff, bs = -1;
        for (int i = tid; i < n; i += blockDim.x) {
            float v = sv[i]; int id = si[i];
            if (v > bv || (v == bv && id < bi)) { bv = v; bi = id; bs = i; }
        }
        rv[tid] = bv; ri[tid] = bi; rs[tid] = bs;
        for (int s = 128; s > 0; s >>= 1) {
            __syncthreads();
            if (tid < s) {
                float ov = rv[tid + s]; int oi = ri[tid + s];
                if (ov > rv[tid] || (ov == rv[tid] && oi < ri[tid])) {
                    rv[tid] = ov; ri[tid] = oi; rs[tid] = rs[tid + s];
                }
            }
        }
        __syncthreads();
        if (tid == 0) {
            float val = rv[0]; int gid = ri[0]; int slot = rs[0];
            int q   = gid / LBL;
            int tok = gid % LBL;
            int cl = 0, lo = 0, hi = 79;
            while (lo <= hi) {
                int mid = (lo + hi) >> 1;
                int v = c_token_idxs[mid];
                if (v == tok) { cl = mid; break; }
                if (v < tok) lo = mid + 1; else hi = mid - 1;
            }
            out[b * k + sel] = val;
            out[BATCH * k + b * k + sel] = (float)cl;
            const float* bb = bbox + ((size_t)b * QRY + q) * 4;
            float cx = bb[0], cy = bb[1], bw = bb[2], bh = bb[3];
            float ih = tsize[b * 2 + 0], iw = tsize[b * 2 + 1];
            size_t base = (size_t)2 * BATCH * k + ((size_t)b * k + sel) * 4;
            out[base + 0] = (cx - 0.5f * bw) * iw;
            out[base + 1] = (cy - 0.5f * bh) * ih;
            out[base + 2] = (cx + 0.5f * bw) * iw;
            out[base + 3] = (cy + 0.5f * bh) * ih;
            if (slot >= 0) sv[slot] = -FLT_MAX;   // remove winner
        }
        __syncthreads();
    }
}

// ---------------- host launch ----------------
extern "C" void kernel_launch(void* const* d_in, const int* in_sizes, int n_in,
                              void* d_out, int out_size, void* d_ws, size_t ws_size,
                              hipStream_t stream) {
    const float* logits  = (const float*)d_in[0];   // (64,900,256)
    const float* bbox    = (const float*)d_in[1];   // (64,900,4)
    const float* pos_map = (const float*)d_in[2];   // (769,256)
    const float* tsize   = (const float*)d_in[3];   // (64,2)
    const int*   nsel    = (const int*)d_in[4];     // scalar (50)
    float* out = (float*)d_out;

    // workspace carving (all offsets 256-aligned)
    char* w = (char*)d_ws;
    size_t off = 0;
    _Float16* A_h  = (_Float16*)(w + off); off += (size_t)BATCH * QPAD * TOK * 2;   // 30.4 MB
    _Float16* B_h  = (_Float16*)(w + off); off += (size_t)LPAD * TOK * 2;           // 0.43 MB
    int*      hist = (int*)(w + off);      off += (size_t)BATCH * HBINS * 4;        // 256 KB
    int*      cntp = (int*)(w + off);      off += 256;
    float*    thr  = (float*)(w + off);    off += 256;
    float*    cV   = (float*)(w + off);    off += (size_t)BATCH * CAP * 4;          // 1 MB
    int*      cI   = (int*)(w + off);      off += (size_t)BATCH * CAP * 4;          // 1 MB
    (void)ws_size; (void)in_sizes; (void)n_in; (void)out_size;

    // 0) zero histogram + counters
    {
        int total = BATCH * HBINS;
        k_zero<<<(total + 255) / 256, 256, 0, stream>>>(hist, cntp);
    }
    // 1) sigmoid + f16 convert (padded)
    {
        int total = BATCH * QPAD * TOK;
        k_prep_a<<<(total + 255) / 256, 256, 0, stream>>>(logits, A_h);
    }
    {
        int total = LPAD * TOK;
        k_prep_b<<<(total + 255) / 256, 256, 0, stream>>>(pos_map, B_h);
    }
    // 2) WMMA GEMM (B in LDS) -> histogram
    {
        dim3 grid(LT * QTG, BATCH);
        k_hist<<<grid, 256, 0, stream>>>(A_h, B_h, hist);
    }
    // 3) threshold per batch
    k_thresh<<<BATCH, 32, 0, stream>>>(hist, nsel, thr);
    // 4) WMMA GEMM (B in LDS) -> candidate collection
    {
        dim3 grid(LT * QTG, BATCH);
        k_collect<<<grid, 256, 0, stream>>>(A_h, B_h, thr, cntp, cV, cI);
    }
    // 5) exact top-k + box/label outputs
    k_select<<<BATCH, 256, 0, stream>>>(cV, cI, cntp, nsel, bbox, tsize, out);
}
